// ResidualLSTMBlock_14482629722209
// MI455X (gfx1250) — compile-verified, matrix operand order checked by reading, not run on hardware
//
#include <hip/hip_runtime.h>
#include <hip/hip_bf16.h>

// Problem constants (from reference)
#define B_    128
#define T_    1024
#define F_    256
#define U_    512
#define FOURU 2048
#define MROWS (B_ * T_)   // 131072
#define NWG_REC 32        // persistent workgroups in the recurrent kernel

typedef __attribute__((ext_vector_type(16))) __bf16 bf16x16;
typedef __attribute__((ext_vector_type(8)))  __bf16 bf16x8;
typedef __attribute__((ext_vector_type(8)))  float  f32x8;

// ---------------------------------------------------------------------------
// WMMA fragment helpers (CDNA5 16-bit A layout, CDNA5 ISA §7.12.2):
//   lane<16  holds K = {0..7} u {16..23}   (element e: k = (e&8)<<1 | (e&7))
//   lane>=16 holds K = {8..15} u {24..31}  (+8)
// B mirrors A with lane = column. C/D: element r -> row r+8*(lane>>4), col lane&15.
// ---------------------------------------------------------------------------
__device__ __forceinline__ int frag_k(int e, int hi) {
  return ((e & 8) << 1) + (e & 7) + (hi << 3);
}

__device__ __forceinline__ bf16x16 cat8(bf16x8 lo, bf16x8 hi) {
  return __builtin_shufflevector(lo, hi, 0, 1, 2, 3, 4, 5, 6, 7,
                                         8, 9, 10, 11, 12, 13, 14, 15);
}

__device__ __forceinline__ f32x8 wmma_bf16(bf16x16 a, bf16x16 b, f32x8 c) {
  return __builtin_amdgcn_wmma_f32_16x16x32_bf16(
      /*neg_a=*/false, a, /*neg_b=*/false, b,
      /*c_mod=*/(short)0, c, /*reuse_a=*/false, /*reuse_b=*/false);
}

__device__ __forceinline__ float sigf(float x) { return 1.0f / (1.0f + __expf(-x)); }

// ---------------------------------------------------------------------------
// Weight f32 -> bf16 conversion
// ---------------------------------------------------------------------------
__global__ void __launch_bounds__(256)
cvt_f32_bf16(const float* __restrict__ s, __bf16* __restrict__ d, int n) {
  int i = blockIdx.x * blockDim.x + threadIdx.x;
  if (i < n) d[i] = (__bf16)s[i];
}

// ---------------------------------------------------------------------------
// Phase 1: out[M,N] = X[M,256] @ W[256,N] + bias   (M = 131072)
// WG = 8 waves = 128 rows x 64 cols. B panel staged to LDS in fragment order.
// OUT_BF16=1 -> xz (bf16, N=2048); OUT_BF16=0 -> residual (f32 to d_out, N=512)
// ---------------------------------------------------------------------------
template <bool OUT_BF16>
__global__ void __launch_bounds__(256)
gemm_x(const float* __restrict__ X, const __bf16* __restrict__ W,
       const float* __restrict__ bias, void* __restrict__ out, int N) {
  __shared__ alignas(32) __bf16 Bp[8][4][32][16];  // 32 KB: [kb][ntile][lane][elem]
  const int tid = threadIdx.x;
  const int n0 = blockIdx.x * 64;
  const int m_base = blockIdx.y * 128;

  // Stage B panel (K=256 x 64 cols) into WMMA fragment order.
  for (int idx = tid; idx < 8 * 4 * 32; idx += 256) {
    const int lane = idx & 31;
    const int nt = (idx >> 5) & 3;
    const int kb = idx >> 7;
    const int hi = lane >> 4;
    const int c = n0 + nt * 16 + (lane & 15);
#pragma unroll
    for (int e = 0; e < 16; ++e) {
      const int k = kb * 32 + frag_k(e, hi);
      Bp[kb][nt][lane][e] = W[(size_t)k * N + c];
    }
  }
  __syncthreads();

  const int wave = tid >> 5, lane = tid & 31;
  const int hi = lane >> 4, rl = lane & 15;
  const int m0 = m_base + wave * 16;

  f32x8 acc0{}, acc1{}, acc2{}, acc3{};
#pragma unroll
  for (int kb = 0; kb < 8; ++kb) {
    // A fragment: row m0+rl, two contiguous 8-float runs, convert f32->bf16.
    const float* ap = X + (size_t)(m0 + rl) * F_ + kb * 32 + hi * 8;
    const float4 x0 = *(const float4*)(ap);
    const float4 x1 = *(const float4*)(ap + 4);
    const float4 x2 = *(const float4*)(ap + 16);
    const float4 x3 = *(const float4*)(ap + 20);
    bf16x16 a;
    a[0] = (__bf16)x0.x; a[1] = (__bf16)x0.y; a[2]  = (__bf16)x0.z; a[3]  = (__bf16)x0.w;
    a[4] = (__bf16)x1.x; a[5] = (__bf16)x1.y; a[6]  = (__bf16)x1.z; a[7]  = (__bf16)x1.w;
    a[8] = (__bf16)x2.x; a[9] = (__bf16)x2.y; a[10] = (__bf16)x2.z; a[11] = (__bf16)x2.w;
    a[12] = (__bf16)x3.x; a[13] = (__bf16)x3.y; a[14] = (__bf16)x3.z; a[15] = (__bf16)x3.w;

    acc0 = wmma_bf16(a, cat8(*(const bf16x8*)&Bp[kb][0][lane][0],
                             *(const bf16x8*)&Bp[kb][0][lane][8]), acc0);
    acc1 = wmma_bf16(a, cat8(*(const bf16x8*)&Bp[kb][1][lane][0],
                             *(const bf16x8*)&Bp[kb][1][lane][8]), acc1);
    acc2 = wmma_bf16(a, cat8(*(const bf16x8*)&Bp[kb][2][lane][0],
                             *(const bf16x8*)&Bp[kb][2][lane][8]), acc2);
    acc3 = wmma_bf16(a, cat8(*(const bf16x8*)&Bp[kb][3][lane][0],
                             *(const bf16x8*)&Bp[kb][3][lane][8]), acc3);
  }

#pragma unroll
  for (int nt = 0; nt < 4; ++nt) {
    const f32x8 acc = (nt == 0) ? acc0 : (nt == 1) ? acc1 : (nt == 2) ? acc2 : acc3;
    const int col = n0 + nt * 16 + rl;
    const float bv = bias[col];
#pragma unroll
    for (int r = 0; r < 8; ++r) {
      const int row = m0 + r + 8 * hi;
      const float v = acc[r] + bv;
      if (OUT_BF16) ((__bf16*)out)[(size_t)row * N + col] = (__bf16)v;
      else          ((float*)out)[(size_t)row * N + col] = v;
    }
  }
}

// ---------------------------------------------------------------------------
// Phase 2: persistent recurrent kernel. 32 WGs x 8 waves; WG owns 16 U-cols,
// wave owns a 16-row batch tile. rec_kernel fragments resident in LDS (64 KB);
// h ping-pongs through a bf16 buffer in L2; c lives in VGPRs for all T steps.
// Device-wide barrier per timestep.
// ---------------------------------------------------------------------------
__global__ void __launch_bounds__(256)
lstm_rec(const __bf16* __restrict__ xz, const float* __restrict__ recw,
         __bf16* __restrict__ hbuf, __bf16* __restrict__ hs,
         unsigned* __restrict__ bar) {
  __shared__ alignas(32) __bf16 Bw[4][16][32][16];  // 64 KB: [gate][kb][lane][elem]
  const int tid = threadIdx.x;
  const int u0 = blockIdx.x * 16;

  // Stage rec_kernel columns {u0..u0+15} for all 4 gates, fragment order, once.
  for (int idx = tid; idx < 4 * 16 * 32; idx += 256) {
    const int lane = idx & 31;
    const int kb = (idx >> 5) & 15;
    const int g = idx >> 9;
    const int hi = lane >> 4;
    const int c = g * U_ + u0 + (lane & 15);
#pragma unroll
    for (int e = 0; e < 16; ++e) {
      const int k = kb * 32 + frag_k(e, hi);
      Bw[g][kb][lane][e] = (__bf16)recw[(size_t)k * FOURU + c];
    }
  }
  __syncthreads();

  const int wave = tid >> 5, lane = tid & 31;
  const int hi = lane >> 4, rl = lane & 15;
  const int m0 = wave * 16;

  f32x8 cstate{};  // cell state, register-resident for all T steps

  for (int t = 0; t < T_; ++t) {
    const __bf16* hread = hbuf + (size_t)(t & 1) * (B_ * U_);
    __bf16* hwrite = hbuf + (size_t)((t + 1) & 1) * (B_ * U_);

    f32x8 acc0{}, acc1{}, acc2{}, acc3{};
#pragma unroll 4
    for (int kb = 0; kb < 16; ++kb) {
      const __bf16* ap = hread + (size_t)(m0 + rl) * U_ + kb * 32 + hi * 8;
      const bf16x16 a = cat8(*(const bf16x8*)(ap), *(const bf16x8*)(ap + 16));
      acc0 = wmma_bf16(a, cat8(*(const bf16x8*)&Bw[0][kb][lane][0],
                               *(const bf16x8*)&Bw[0][kb][lane][8]), acc0);
      acc1 = wmma_bf16(a, cat8(*(const bf16x8*)&Bw[1][kb][lane][0],
                               *(const bf16x8*)&Bw[1][kb][lane][8]), acc1);
      acc2 = wmma_bf16(a, cat8(*(const bf16x8*)&Bw[2][kb][lane][0],
                               *(const bf16x8*)&Bw[2][kb][lane][8]), acc2);
      acc3 = wmma_bf16(a, cat8(*(const bf16x8*)&Bw[3][kb][lane][0],
                               *(const bf16x8*)&Bw[3][kb][lane][8]), acc3);
    }

#pragma unroll
    for (int r = 0; r < 8; ++r) {
      const int brow = m0 + r + 8 * hi;  // batch index
      const size_t xb = ((size_t)brow * T_ + t) * FOURU + u0 + rl;
      const float zi = acc0[r] + (float)xz[xb];
      const float zf = acc1[r] + (float)xz[xb + U_];
      const float zg = acc2[r] + (float)xz[xb + 2 * U_];
      const float zo = acc3[r] + (float)xz[xb + 3 * U_];
      const float ig = sigf(zi), fg = sigf(zf), og = sigf(zo);
      const float cv = fg * cstate[r] + ig * tanhf(zg);
      cstate[r] = cv;
      const float hv = og * tanhf(cv);
      hwrite[(size_t)brow * U_ + u0 + rl] = (__bf16)hv;
      hs[((size_t)brow * T_ + t) * U_ + u0 + rl] = (__bf16)hv;
    }

    // ---- device-wide barrier (counter + generation) ----
    __threadfence();
    __syncthreads();
    if (tid == 0) {
      unsigned g0 = __hip_atomic_load(&bar[1], __ATOMIC_ACQUIRE, __HIP_MEMORY_SCOPE_AGENT);
      unsigned a = __hip_atomic_fetch_add(&bar[0], 1u, __ATOMIC_ACQ_REL, __HIP_MEMORY_SCOPE_AGENT);
      if (a == (unsigned)(NWG_REC - 1)) {
        __hip_atomic_store(&bar[0], 0u, __ATOMIC_RELAXED, __HIP_MEMORY_SCOPE_AGENT);
        __hip_atomic_fetch_add(&bar[1], 1u, __ATOMIC_RELEASE, __HIP_MEMORY_SCOPE_AGENT);
      } else {
        while (__hip_atomic_load(&bar[1], __ATOMIC_ACQUIRE, __HIP_MEMORY_SCOPE_AGENT) == g0) {
          __builtin_amdgcn_s_sleep(2);
        }
      }
    }
    __syncthreads();
    __threadfence();
  }
}

// ---------------------------------------------------------------------------
// Phase 3: y = hs + residual (already in d_out); LayerNorm over U=512 in place.
// One block per (b,t) row; 256 threads x 2 elements.
// ---------------------------------------------------------------------------
__global__ void __launch_bounds__(256)
finalize_ln(const __bf16* __restrict__ hs, const float* __restrict__ gamma,
            const float* __restrict__ beta, float* __restrict__ out) {
  __shared__ float ssum[256], sq[256];
  const size_t base = (size_t)blockIdx.x * U_;
  const int tid = threadIdx.x;
  const float y0 = (float)hs[base + tid] + out[base + tid];
  const float y1 = (float)hs[base + 256 + tid] + out[base + 256 + tid];
  ssum[tid] = y0 + y1;
  sq[tid] = y0 * y0 + y1 * y1;
  __syncthreads();
  for (int s = 128; s > 0; s >>= 1) {
    if (tid < s) { ssum[tid] += ssum[tid + s]; sq[tid] += sq[tid + s]; }
    __syncthreads();
  }
  const float mu = ssum[0] * (1.0f / U_);
  const float var = sq[0] * (1.0f / U_) - mu * mu;
  const float rs = rsqrtf(var + 1e-6f);
  out[base + tid] = gamma[tid] * (y0 - mu) * rs + beta[tid];
  out[base + 256 + tid] = gamma[tid + 256] * (y1 - mu) * rs + beta[tid + 256];
}

// ---------------------------------------------------------------------------
extern "C" void kernel_launch(void* const* d_in, const int* in_sizes, int n_in,
                              void* d_out, int out_size, void* d_ws, size_t ws_size,
                              hipStream_t stream) {
  (void)in_sizes; (void)n_in; (void)out_size; (void)ws_size;
  const float* x       = (const float*)d_in[0];
  const float* kernelw = (const float*)d_in[1];
  const float* recw    = (const float*)d_in[2];
  const float* bias    = (const float*)d_in[3];
  const float* projw   = (const float*)d_in[4];
  const float* projb   = (const float*)d_in[5];
  const float* gamma   = (const float*)d_in[6];
  const float* beta    = (const float*)d_in[7];
  float* out = (float*)d_out;

  // Workspace layout (all offsets 256B-aligned): ~642 MB total.
  char* ws = (char*)d_ws;
  unsigned* bar = (unsigned*)ws;                                  //      256 B
  __bf16* hbuf = (__bf16*)(ws + 256);                             //  262144 B (h ping-pong)
  size_t off = 256 + 262144;
  __bf16* xzbf = (__bf16*)(ws + off); off += (size_t)MROWS * FOURU * 2;  // 512 MB
  __bf16* hsbf = (__bf16*)(ws + off); off += (size_t)MROWS * U_ * 2;     // 128 MB
  __bf16* kbf  = (__bf16*)(ws + off); off += (size_t)F_ * FOURU * 2;     //   1 MB
  __bf16* pbf  = (__bf16*)(ws + off);                                    // 256 KB

  // Zero barrier + h ping-pong every call (deterministic; capture-safe).
  hipMemsetAsync(ws, 0, 256 + 262144, stream);

  { int n = F_ * FOURU; cvt_f32_bf16<<<(n + 255) / 256, 256, 0, stream>>>(kernelw, kbf, n); }
  { int n = F_ * U_;    cvt_f32_bf16<<<(n + 255) / 256, 256, 0, stream>>>(projw, pbf, n); }

  // xz = x@kernel + bias -> bf16 ws ; residual = x@proj_w + proj_b -> f32 d_out
  gemm_x<true ><<<dim3(FOURU / 64, MROWS / 128), 256, 0, stream>>>(x, kbf, bias, xzbf, FOURU);
  gemm_x<false><<<dim3(U_ / 64,    MROWS / 128), 256, 0, stream>>>(x, pbf, projb, out, U_);

  // Sequential scan: persistent kernel, one device barrier per timestep.
  lstm_rec<<<NWG_REC, 256, 0, stream>>>(xzbf, recw, hbuf, hsbf, bar);

  // y = hs + residual, LayerNorm in place.
  finalize_ln<<<MROWS, 256, 0, stream>>>(hsbf, gamma, beta, out);
}
